// PMPGNN_75797582840356
// MI455X (gfx1250) — matrix operand order repaired and checked by reading, not compiled, not run on hardware
//
#include <hip/hip_runtime.h>
#include <hip/hip_bf16.h>
#include <math.h>

// ---------------- types for WMMA ----------------
typedef __attribute__((ext_vector_type(8)))  float  v8f;
typedef __attribute__((ext_vector_type(16))) __bf16 v16bf;

union Frag { unsigned u[8]; v16bf v; };

// f32 -> bf16 (RNE) pack of two consecutive values into one dword
__device__ __forceinline__ unsigned short f2bf(float x) {
  unsigned u = __float_as_uint(x);
  unsigned r = (u + 0x7FFFu + ((u >> 16) & 1u)) >> 16;
  return (unsigned short)r;
}
__device__ __forceinline__ unsigned packbf(float lo, float hi) {
  return (unsigned)f2bf(lo) | ((unsigned)f2bf(hi) << 16);
}

__device__ __forceinline__ float waveRed32(float v) {
#pragma unroll
  for (int m = 16; m >= 1; m >>= 1) v += __shfl_xor(v, m, 32);
  return v;
}
__device__ __forceinline__ float red16(float v) {
#pragma unroll
  for (int m = 8; m >= 1; m >>= 1) v += __shfl_xor(v, m, 16);
  return v;
}

// ---------------- kernels ----------------

__global__ void kzero(float* __restrict__ p, long n) {
  long t = (long)blockIdx.x * blockDim.x + threadIdx.x;
  if (t < n) p[t] = 0.f;
}

// out = elu(in @ W + b) for 16x16 W; also sw = out * ws[c]
__global__ void kmlp16(const float* __restrict__ in, const float* __restrict__ W,
                       const float* __restrict__ bias, const float* __restrict__ ws,
                       float* __restrict__ out, float* __restrict__ swout, int n) {
  long t = (long)blockIdx.x * blockDim.x + threadIdx.x;
  if (t >= (long)n * 16) return;
  int node = (int)(t >> 4), c = (int)(t & 15);
  const float* row = in + (long)node * 16;
  float acc = bias[c];
#pragma unroll
  for (int k = 0; k < 16; ++k) acc += row[k] * W[k * 16 + c];
  acc = acc > 0.f ? acc : expf(acc) - 1.f;
  out[t] = acc;
  swout[t] = acc * ws[c];
}

// Pack f32 row-major KxN weight matrix into bf16 WMMA B-fragments:
// out[((nt*KT + kt)*8 + j)*32 + lane]; B layout: VGPR j, lane = half*16+n,
// holds B[k = kt*32 + (j&4)*4 + half*8 + (j&3)*2][n] (lo) and k+1 (hi).
__global__ void kpackB(const float* __restrict__ B, int K, int Ncols,
                       unsigned* __restrict__ out) {
  int t = blockIdx.x * blockDim.x + threadIdx.x;
  int KT = K >> 5;
  int total = (Ncols >> 4) * KT * 8 * 32;
  if (t >= total) return;
  int lane = t & 31;
  int j = (t >> 5) & 7;
  int kt = (t >> 8) % KT;
  int nt = (t >> 8) / KT;
  int half = lane >> 4;
  int nn = nt * 16 + (lane & 15);
  int k = kt * 32 + ((j & 4) << 2) + half * 8 + ((j & 3) << 1);
  out[t] = packbf(B[(long)k * Ncols + nn], B[(long)(k + 1) * Ncols + nn]);
}

// Pack f32 row-major MxK matrix into bf16 WMMA A-fragments:
// out[((mt*KT + kt)*8 + j)*32 + lane]; A layout: lane = half*16+m,
// VGPR j holds K = kt*32 + (j&4)*4 + half*8 + (j&3)*2, +1.
__global__ void kpackA(const float* __restrict__ A, int K, int KT,
                       unsigned* __restrict__ out, long total) {
  long t = (long)blockIdx.x * blockDim.x + threadIdx.x;
  if (t >= total) return;
  int lane = (int)(t & 31);
  int j = (int)(t >> 5) & 7;
  long grp = t >> 8;                 // mt*KT + kt
  int kt = (int)(grp % KT);
  long mt = grp / KT;
  int half = lane >> 4;
  long row = mt * 16 + (lane & 15);
  int k = kt * 32 + ((j & 4) << 2) + half * 8 + ((j & 3) << 1);
  const float* p = A + row * K + k;
  out[t] = packbf(p[0], p[1]);
}

// Packed-A GEMM: C(MxNcols, f32) = Ap @ Bp, TN 16x16 N-tiles per wave.
// Inner loop: coalesced dword fragment loads + TN WMMAs per 32-deep K step.
template <int TN>
__global__ void kgemmp(const unsigned* __restrict__ Ap, const unsigned* __restrict__ Bp,
                       float* __restrict__ C, int MT, int NT, int KT, int Ncols) {
  int wid  = (blockIdx.x * blockDim.x + threadIdx.x) >> 5;
  int lane = threadIdx.x & 31;
  int ngrp = NT / TN;
  if (wid >= MT * ngrp) return;          // wave-uniform: EXEC stays all-ones
  int mt = wid / ngrp, ng = wid % ngrp;
  const unsigned* abase = Ap + (long)mt * KT * 256 + lane;
  const unsigned* bbase[TN];
#pragma unroll
  for (int tn = 0; tn < TN; ++tn)
    bbase[tn] = Bp + (long)(ng * TN + tn) * KT * 256 + lane;
  v8f acc[TN] = {};
  for (int kt = 0; kt < KT; ++kt) {
    if (kt + 1 < KT) {                   // prefetch next K-step fragments
      __builtin_prefetch(abase + (kt + 1) * 256, 0, 3);
      __builtin_prefetch(bbase[0] + (kt + 1) * 256, 0, 3);
    }
    Frag a;
#pragma unroll
    for (int j = 0; j < 8; ++j) a.u[j] = abase[(kt * 8 + j) * 32];
#pragma unroll
    for (int tn = 0; tn < TN; ++tn) {
      Frag b;
#pragma unroll
      for (int j = 0; j < 8; ++j) b.u[j] = bbase[tn][(kt * 8 + j) * 32];
      acc[tn] = __builtin_amdgcn_wmma_f32_16x16x32_bf16(false, a.v, false, b.v,
                                                        (short)0, acc[tn], false, false);
    }
  }
  int half = lane >> 4, q = lane & 15;
#pragma unroll
  for (int tn = 0; tn < TN; ++tn) {
    int n = (ng * TN + tn) * 16 + q;
#pragma unroll
    for (int r = 0; r < 8; ++r) {        // C/D: vgpr r -> M = half*8 + r, N = q
      int m = mt * 16 + half * 8 + r;
      C[(long)m * Ncols + n] = acc[tn][r];
    }
  }
}

// Fallback GEMM (A loaded from f32 and converted in-loop) if scratch is small.
__global__ void kgemm(const float* __restrict__ A, int K,
                      const unsigned* __restrict__ Bp,
                      float* __restrict__ C, int MT, int NT, int KT, int Ncols) {
  int wid  = (blockIdx.x * blockDim.x + threadIdx.x) >> 5;
  int lane = threadIdx.x & 31;
  if (wid >= MT * NT) return;
  int mt = wid / NT, nt = wid % NT;
  int half = lane >> 4, q = lane & 15;
  const float*    arow  = A + (long)(mt * 16 + q) * K;
  const unsigned* bbase = Bp + (long)nt * KT * 8 * 32 + lane;
  v8f acc = {};
  for (int kt = 0; kt < KT; ++kt) {
    Frag a, b;
    int k0 = kt * 32;
#pragma unroll
    for (int j = 0; j < 8; ++j) {
      int k = k0 + ((j & 4) << 2) + half * 8 + ((j & 3) << 1);
      float2 p = *(const float2*)(arow + k);
      a.u[j] = packbf(p.x, p.y);
      b.u[j] = bbase[(kt * 8 + j) * 32];
    }
    acc = __builtin_amdgcn_wmma_f32_16x16x32_bf16(false, a.v, false, b.v,
                                                  (short)0, acc, false, false);
  }
  int n = nt * 16 + q;
#pragma unroll
  for (int r = 0; r < 8; ++r) {
    int m = mt * 16 + half * 8 + r;
    C[(long)m * Ncols + n] = acc[r];
  }
}

__global__ void kdeg(const int* __restrict__ dst, float* __restrict__ indeg, int E) {
  int t = blockIdx.x * blockDim.x + threadIdx.x;
  if (t < E) atomicAdd(&indeg[dst[t]], 1.0f);
}

// per-node: lr = leakyrelu(h,0.2); el = lr.wu, er = lr.wv  (wave per node)
__global__ void knode(const float* __restrict__ h, int F,
                      const float* __restrict__ wu, const float* __restrict__ wv,
                      float* __restrict__ el, float* __restrict__ er, int n) {
  int wid  = (blockIdx.x * blockDim.x + threadIdx.x) >> 5;
  int lane = threadIdx.x & 31;
  if (wid >= n) return;
  const float* row = h + (long)wid * F;
  float a = 0.f, b = 0.f;
  for (int k = lane; k < F; k += 32) {
    float v = row[k];
    v = v > 0.f ? v : 0.2f * v;
    a += v * wu[k];
    b += v * wv[k];
  }
  a = waveRed32(a);
  b = waveRed32(b);
  if (lane == 0) { el[wid] = a; er[wid] = b; }
}

// per-edge weight: ew = exp(e - betaw*(w0*sdf + w1*sds)) + eps; scatter to out/in sums
__global__ void kedge(const int* __restrict__ src, const int* __restrict__ dst,
                      const float* __restrict__ h, int F,
                      const float* __restrict__ el, const float* __restrict__ er,
                      const float* __restrict__ sw, const float* __restrict__ sf,
                      const float* __restrict__ beta, const float* __restrict__ aw,
                      float* __restrict__ ew, float* __restrict__ outw,
                      float* __restrict__ inw, int E) {
  int wid  = (blockIdx.x * blockDim.x + threadIdx.x) >> 5;
  int lane = threadIdx.x & 31;
  if (wid >= E) return;
  int s = src[wid], d = dst[wid];
  const float* hs = h + (long)s * F;
  const float* hd = h + (long)d * F;
  float sdf = 0.f;
  for (int k = lane; k < F; k += 32) {
    float df = hs[k] - hd[k];
    sdf += df * df;
  }
  float st = 0.f, sds = 0.f;
  if (lane < 16) {
    float b2 = sf[(long)d * 16 + lane];
    st = sw[(long)s * 16 + lane] * b2;
    float df = sf[(long)s * 16 + lane] - b2;
    sds = df * df;
  }
  sdf = waveRed32(sdf);
  st  = waveRed32(st);
  sds = waveRed32(sds);
  if (lane == 0) {
    float e0 = expf(aw[0]), e1 = expf(aw[1]);
    float w0 = e0 / (e0 + e1), w1 = 1.f - w0;
    float betaw = 2.f / (expf(-beta[0]) + 1.f);
    float e = el[s] + er[d] + st;
    float v = expf(e - betaw * (w0 * sdf + w1 * sds)) + 1e-9f;
    ew[wid] = v;
    atomicAdd(&outw[s], v);
    atomicAdd(&inw[d], v);
  }
}

// rst[dst] += h[src] * (outw[src]^-.5 * inw[dst]^-.5 * ew)
__global__ void kscatter(const int* __restrict__ src, const int* __restrict__ dst,
                         const float* __restrict__ h, int F,
                         const float* __restrict__ ew, const float* __restrict__ outw,
                         const float* __restrict__ inw, float* __restrict__ rst, int E) {
  int wid  = (blockIdx.x * blockDim.x + threadIdx.x) >> 5;
  int lane = threadIdx.x & 31;
  if (wid >= E) return;
  int s = src[wid], d = dst[wid];
  float a = ew[wid] * rsqrtf(outw[s]) * rsqrtf(inw[d]);
  const float* hs = h + (long)s * F;
  float* rd = rst + (long)d * F;
  for (int k = lane; k < F; k += 32) atomicAdd(&rd[k], hs[k] * a);
}

// layer-1 tail: rst = elu(rst + h*perm), in place (F=128)
__global__ void kfinal1(float* __restrict__ rst, const float* __restrict__ h,
                        const float* __restrict__ indeg,
                        const float* __restrict__ theta, int n) {
  long t = (long)blockIdx.x * blockDim.x + threadIdx.x;
  if (t >= (long)n * 128) return;
  int node = (int)(t >> 7);
  float ps = 1e-9f / (expf(-theta[0]) + 1.f);
  float perm = ps / (indeg[node] + 1e-9f);
  float v = rst[t] + h[t] * perm;
  rst[t] = v > 0.f ? v : expf(v) - 1.f;
}

// layer-2 tail + log_softmax over 16 classes (wave per node, lanes 0..15 active data)
__global__ void kfinal2(const float* __restrict__ rst2, const float* __restrict__ h2,
                        const float* __restrict__ indeg, const float* __restrict__ theta,
                        float* __restrict__ out, int n) {
  int wid  = (blockIdx.x * blockDim.x + threadIdx.x) >> 5;
  int lane = threadIdx.x & 31;
  if (wid >= n) return;
  float ps = 1e-9f / (expf(-theta[0]) + 1.f);
  float perm = ps / (indeg[wid] + 1e-9f);
  float v = 0.f;
  if (lane < 16) {
    float x = rst2[(long)wid * 16 + lane] + h2[(long)wid * 16 + lane] * perm;
    v = x > 0.f ? x : expf(x) - 1.f;
  }
  float mx = v;
#pragma unroll
  for (int m = 8; m >= 1; m >>= 1) mx = fmaxf(mx, __shfl_xor(mx, m, 16));
  float ex = (lane < 16) ? expf(v - mx) : 0.f;
  float se = red16(ex);
  if (lane < 16) out[(long)wid * 16 + lane] = (v - mx) - logf(se);
}

// lp_loss = sum_e w[e] * <hl2[src], hl2[dst]>; LDS block reduction + one atomic
__global__ void klploss(const int* __restrict__ src, const int* __restrict__ dst,
                        const float* __restrict__ w, const float* __restrict__ hl,
                        float* __restrict__ loss, int E) {
  __shared__ float smem[8];
  long t = (long)blockIdx.x * blockDim.x + threadIdx.x;
  float p = 0.f;
  if (t < (long)E * 16) {
    int e = (int)(t >> 4), c = (int)(t & 15);
    p = w[e] * hl[(long)src[e] * 16 + c] * hl[(long)dst[e] * 16 + c];
  }
  p = red16(p);
  p += __shfl_xor(p, 16, 32);
  int lane = threadIdx.x & 31;
  if (lane == 0) smem[threadIdx.x >> 5] = p;
  __syncthreads();
  if (threadIdx.x == 0) {
    float sum = 0.f;
#pragma unroll
    for (int i = 0; i < 8; ++i) sum += smem[i];
    atomicAdd(loss, sum);
  }
}

// ---------------- launch ----------------

extern "C" void kernel_launch(void* const* d_in, const int* in_sizes, int n_in,
                              void* d_out, int out_size, void* d_ws, size_t ws_size,
                              hipStream_t stream) {
  constexpr int N = 50000, E = 800000, F1 = 128, NC = 16, K1 = 512;
  constexpr int MT = N / 16;          // 3125 (exact)
  constexpr int KT1 = K1 / 32;        // 16
  constexpr int NT1 = F1 / 16;        // 8
  constexpr int KT2 = F1 / 32;        // 4
  constexpr int NT2 = NC / 16;        // 1

  const int*   src     = (const int*)d_in[0];
  const int*   dst     = (const int*)d_in[1];
  const float* x       = (const float*)d_in[2];
  const float* weights = (const float*)d_in[3];
  const float* latp    = (const float*)d_in[4];
  const float* sl_w    = (const float*)d_in[5];
  const float* sl_b    = (const float*)d_in[6];
  const float* slo_w   = (const float*)d_in[7];
  const float* slo_b   = (const float*)d_in[8];
  const float* fc1_w   = (const float*)d_in[9];
  const float* wu1     = (const float*)d_in[10];
  const float* wv1     = (const float*)d_in[11];
  const float* ws1     = (const float*)d_in[12];
  const float* beta1   = (const float*)d_in[13];
  const float* aw1     = (const float*)d_in[14];
  const float* theta1  = (const float*)d_in[15];
  const float* fc2_w   = (const float*)d_in[16];
  const float* wu2     = (const float*)d_in[17];
  const float* wv2     = (const float*)d_in[18];
  const float* ws2     = (const float*)d_in[19];
  const float* beta2   = (const float*)d_in[20];
  const float* aw2     = (const float*)d_in[21];
  const float* theta2  = (const float*)d_in[22];
  float* out = (float*)d_out;

  // workspace layout; first block (indeg..rst1) is contiguously zeroed each call
  float* w = (float*)d_ws;
  float* indeg = w; w += N;
  float* outw1 = w; w += N;
  float* inw1  = w; w += N;
  float* outw2 = w; w += N;
  float* inw2  = w; w += N;
  float* rst2  = w; w += (size_t)N * NC;
  float* rst1  = w; w += (size_t)N * F1;
  const long zcount = (long)N * (5 + NC + F1);
  float* hlatp  = w; w += (size_t)N * NC;
  float* sw1    = w; w += (size_t)N * NC;
  float* hlatp2 = w; w += (size_t)N * NC;
  float* sw2    = w; w += (size_t)N * NC;
  float* h1     = w; w += (size_t)N * F1;
  float* h2     = w; w += (size_t)N * NC;
  float* el1 = w; w += N;
  float* er1 = w; w += N;
  float* el2 = w; w += N;
  float* er2 = w; w += N;
  float* ew1 = w; w += E;
  float* ew2 = w; w += E;
  unsigned* w1p = (unsigned*)w; w += NT1 * KT1 * 8 * 32;   // 32768 dwords
  unsigned* w2p = (unsigned*)w; w += NT2 * KT2 * 8 * 32;   // 1024 dwords
  // packed-A fragment buffers (used only if scratch is large enough)
  const long xapN  = (long)MT * KT1 * 256;                 // 12.8M dwords
  const long h1apN = (long)MT * KT2 * 256;                 //  3.2M dwords
  unsigned* xap  = (unsigned*)w; w += xapN;
  unsigned* h1ap = (unsigned*)w; w += h1apN;
  const size_t need = (size_t)((char*)w - (char*)d_ws);
  const bool usePacked = need <= ws_size;                  // deterministic per-run

  const int TB = 256;

  // 0) zero accumulation buffers + loss slot (graph-replay safe)
  kzero<<<(int)((zcount + TB - 1) / TB), TB, 0, stream>>>(indeg, zcount);
  kzero<<<1, 32, 0, stream>>>(out + (long)N * NC, 1);

  // 1) tiny MLPs (structure features)
  kmlp16<<<(N * 16 + TB - 1) / TB, TB, 0, stream>>>(latp, sl_w, sl_b, ws1, hlatp, sw1, N);
  kmlp16<<<(N * 16 + TB - 1) / TB, TB, 0, stream>>>(hlatp, slo_w, slo_b, ws2, hlatp2, sw2, N);

  // 2) pack weights into bf16 WMMA B-fragments
  kpackB<<<(NT1 * KT1 * 8 * 32 + TB - 1) / TB, TB, 0, stream>>>(fc1_w, K1, F1, w1p);
  kpackB<<<(NT2 * KT2 * 8 * 32 + TB - 1) / TB, TB, 0, stream>>>(fc2_w, F1, NC, w2p);

  // 3) h1 = x @ fc1_w via v_wmma_f32_16x16x32_bf16
  if (usePacked) {
    kpackA<<<(int)((xapN + TB - 1) / TB), TB, 0, stream>>>(x, K1, KT1, xap, xapN);
    int waves = MT * (NT1 / 4);
    kgemmp<4><<<(waves * 32 + TB - 1) / TB, TB, 0, stream>>>(xap, w1p, h1, MT, NT1, KT1, F1);
  } else {
    kgemm<<<(MT * NT1 * 32 + TB - 1) / TB, TB, 0, stream>>>(x, K1, w1p, h1, MT, NT1, KT1, F1);
  }

  // 4) in-degrees (shared by both layers)
  kdeg<<<(E + TB - 1) / TB, TB, 0, stream>>>(dst, indeg, E);

  // 5) layer-1 attention + aggregation
  knode<<<(N * 32 + TB - 1) / TB, TB, 0, stream>>>(h1, F1, wu1, wv1, el1, er1, N);
  kedge<<<(E * 32 + TB - 1) / TB, TB, 0, stream>>>(src, dst, h1, F1, el1, er1, sw1, hlatp,
                                                   beta1, aw1, ew1, outw1, inw1, E);
  kscatter<<<(E * 32 + TB - 1) / TB, TB, 0, stream>>>(src, dst, h1, F1, ew1, outw1, inw1,
                                                      rst1, E);
  kfinal1<<<(int)(((long)N * F1 + TB - 1) / TB), TB, 0, stream>>>(rst1, h1, indeg, theta1, N);

  // 6) h2 = h1out @ fc2_w via WMMA
  if (usePacked) {
    kpackA<<<(int)((h1apN + TB - 1) / TB), TB, 0, stream>>>(rst1, F1, KT2, h1ap, h1apN);
    kgemmp<1><<<(MT * 32 + TB - 1) / TB, TB, 0, stream>>>(h1ap, w2p, h2, MT, NT2, KT2, NC);
  } else {
    kgemm<<<(MT * NT2 * 32 + TB - 1) / TB, TB, 0, stream>>>(rst1, F1, w2p, h2, MT, NT2, KT2, NC);
  }

  // 7) layer-2 attention + aggregation
  knode<<<(N * 32 + TB - 1) / TB, TB, 0, stream>>>(h2, NC, wu2, wv2, el2, er2, N);
  kedge<<<(E * 32 + TB - 1) / TB, TB, 0, stream>>>(src, dst, h2, NC, el2, er2, sw2, hlatp2,
                                                   beta2, aw2, ew2, outw2, inw2, E);
  kscatter<<<(E * 32 + TB - 1) / TB, TB, 0, stream>>>(src, dst, h2, NC, ew2, outw2, inw2,
                                                      rst2, E);

  // 8) output: log_softmax(elu(rst2 + perm*h2)) and lp_loss
  kfinal2<<<(N * 32 + TB - 1) / TB, TB, 0, stream>>>(rst2, h2, indeg, theta2, out, N);
  klploss<<<(int)(((long)E * 16 + TB - 1) / TB), TB, 0, stream>>>(src, dst, weights, hlatp2,
                                                                  out + (long)N * NC, E);
}
